// MultiHeadAttentionWithRoPE_15642270892196
// MI455X (gfx1250) — matrix operand order repaired and checked by reading, not compiled, not run on hardware
//
#include <hip/hip_runtime.h>
#include <hip/hip_bf16.h>
#include <math.h>

typedef __attribute__((ext_vector_type(16))) _Float16 v16h;
typedef __attribute__((ext_vector_type(8)))  float    v8f;

// Problem constants
constexpr int Bc = 4, Sc = 2048, Dm = 512, Hh = 8, DHc = 64;
constexpr int HD = Hh * DHc;   // 512
constexpr int Mrows = Bc * Sc; // 8192

// ---------------------------------------------------------------------------
// WMMA helpers (16x16x32 f16 -> f32), per CDNA5 ISA VGPR layouts.
// ---------------------------------------------------------------------------
__device__ __forceinline__ v8f wmma_f16(v16h a, v16h b, v8f c) {
  return __builtin_amdgcn_wmma_f32_16x16x32_f16(false, a, false, b,
                                                (short)0, c, false, false);
}

// A fragment from row-major tile: element (m,k) = base[m*ld + k].
// Lane<16 holds K {0..7}+{16..23}, lane>=16 holds K {8..15}+{24..31}:
// two contiguous 16-byte runs -> 2x ds_load_b128. ld must be a multiple of 8.
__device__ __forceinline__ v16h load_a_frag(const _Float16* base, int ld) {
  int lane = threadIdx.x & 31;
  int row  = lane & 15;
  int hi8  = (lane >> 4) << 3;  // 0 or 8
  const _Float16* p = base + row * ld + hi8;
  v16h r;
  *(int4*)&r       = *(const int4*)p;         // K = hi8 + 0..7
  *((int4*)&r + 1) = *(const int4*)(p + 16);  // K = hi8 + 16..23
  return r;
}

// B fragment from N-major (transposed) tile: element (k,n) = base[n*ld + k].
// Lane<16 holds K {0..7}+{16..23}, lane>=16 holds K {8..15}+{24..31} -> 2x ds_load_b128. ld % 8 == 0.
__device__ __forceinline__ v16h load_b_frag_nmajor(const _Float16* base, int ld) {
  int lane = threadIdx.x & 31;
  int n    = lane & 15;
  int khi  = (lane >> 4) << 3;  // 0 or 8
  const _Float16* p = base + n * ld + khi;
  v16h r;
  *(int4*)&r       = *(const int4*)p;
  *((int4*)&r + 1) = *(const int4*)(p + 16);
  return r;
}

// C/D fragment scatter: element row = vgpr + 8*hi, col = lane&15
__device__ __forceinline__ void store_c_frag(float* dst, int ld, v8f c) {
  int lane = threadIdx.x & 31;
  int col  = lane & 15;
  int ro   = (lane >> 4) << 3;
#pragma unroll
  for (int r = 0; r < 8; ++r) dst[(ro + r) * ld + col] = c[r];
}

// ---------------------------------------------------------------------------
// Async global -> LDS staging (GLOBAL_LOAD_ASYNC_TO_LDS_B128, ASYNCcnt).
// Generic LDS pointers carry the LDS byte offset in their low 32 bits.
// ---------------------------------------------------------------------------
__device__ __forceinline__ void async_copy8h(_Float16* dst,
                                             const _Float16* src) {
  unsigned l = (unsigned)(uintptr_t)dst;
  asm volatile("global_load_async_to_lds_b128 %0, %1, off"
               :: "v"(l), "v"(src) : "memory");
}
__device__ __forceinline__ void async_copy16h(_Float16* dst,
                                              const _Float16* src) {
  async_copy8h(dst, src);
  async_copy8h(dst + 8, src + 8);
}
__device__ __forceinline__ void async_wait() {
  asm volatile("s_wait_asynccnt 0x0" ::: "memory");
}

// ---------------------------------------------------------------------------
// f32 -> f16 conversion (for x)
// ---------------------------------------------------------------------------
__global__ void cvt_f32_f16_kernel(const float* __restrict__ src,
                                   _Float16* __restrict__ dst, int n) {
  int i = blockIdx.x * blockDim.x + threadIdx.x;
  if (i < n) dst[i] = (_Float16)src[i];
}

// ---------------------------------------------------------------------------
// f32 [K][N] -> f16 transposed [N][K] (for weights). 32x32 LDS tiles.
// grid = (N/32, K/32), block = 256 (32x8).
// ---------------------------------------------------------------------------
__global__ __launch_bounds__(256) void cvt_transpose_kernel(
    const float* __restrict__ src, _Float16* __restrict__ dst) {
  __shared__ _Float16 T[32][33];
  int n0 = blockIdx.x * 32, k0 = blockIdx.y * 32;
  int tx = threadIdx.x & 31, ty = threadIdx.x >> 5;  // 32 x 8
#pragma unroll
  for (int r = 0; r < 4; ++r) {
    int k = ty * 4 + r;
    T[k][tx] = (_Float16)src[(size_t)(k0 + k) * Dm + n0 + tx];
  }
  __syncthreads();
#pragma unroll
  for (int r = 0; r < 4; ++r) {
    int n = ty * 4 + r;
    dst[(size_t)(n0 + n) * Dm + k0 + tx] = T[tx][n];
  }
}

// ---------------------------------------------------------------------------
// QKV projection (WMMA 128x64 tile) + bias + RoPE + 1/sqrt(DH) folding.
// Weights are pre-transposed f16 [N][K]. V written transposed [B,H,DH,S].
// grid = (64, 8, 3): x = M-tile, y = head, z = {Q,K,V}
// ---------------------------------------------------------------------------
__global__ __launch_bounds__(256) void qkv_rope_kernel(
    const _Float16* __restrict__ xh,
    const _Float16* __restrict__ Wqt, const _Float16* __restrict__ Wkt,
    const _Float16* __restrict__ Wvt,
    const float* __restrict__ bq, const float* __restrict__ bk,
    const float* __restrict__ bv, const float* __restrict__ ts,
    _Float16* __restrict__ Qh, _Float16* __restrict__ Kh,
    _Float16* __restrict__ Vt) {
  __shared__ _Float16 As[128 * 40];    // row-major M x K tile
  __shared__ _Float16 Bs[64 * 40];     // N-major: [n][k]
  __shared__ float    Cs[128 * 66];

  const int z   = blockIdx.z;
  const _Float16* Wt   = (z == 0) ? Wqt : (z == 1) ? Wkt : Wvt;
  const float*    bias = (z == 0) ? bq : (z == 1) ? bk : bv;
  const int m0  = blockIdx.x * 128;
  const int h   = blockIdx.y;
  const int n0h = h * DHc;
  const int tid = threadIdx.x;
  const int wid = tid >> 5;
  const int mBase = (wid & 3) * 32, nBase = (wid >> 2) * 32;

  v8f c[2][2];
#pragma unroll
  for (int i = 0; i < 2; ++i)
#pragma unroll
    for (int j = 0; j < 2; ++j) c[i][j] = (v8f){};

  for (int kk = 0; kk < Dm; kk += 32) {
    {  // stage A tile 128x32 (row-major), async direct-to-LDS
      int row = tid >> 1, seg = (tid & 1) * 16;
      async_copy16h(As + row * 40 + seg,
                    xh + (size_t)(m0 + row) * Dm + kk + seg);
      if (kk + 32 < Dm)
        __builtin_prefetch(xh + (size_t)(m0 + row) * Dm + kk + 32 + seg, 0, 0);
    }
    {  // stage B tile, N-major 64 x 32, async direct-to-LDS
      int n = tid >> 2, seg = (tid & 3) * 8;
      async_copy8h(Bs + n * 40 + seg, Wt + (size_t)(n0h + n) * Dm + kk + seg);
    }
    async_wait();
    __syncthreads();
    v16h a[2], b[2];
#pragma unroll
    for (int i = 0; i < 2; ++i) a[i] = load_a_frag(As + (mBase + 16 * i) * 40, 40);
#pragma unroll
    for (int j = 0; j < 2; ++j) b[j] = load_b_frag_nmajor(Bs + (nBase + 16 * j) * 40, 40);
#pragma unroll
    for (int i = 0; i < 2; ++i)
#pragma unroll
      for (int j = 0; j < 2; ++j) c[i][j] = wmma_f16(a[i], b[j], c[i][j]);
    __syncthreads();
  }

#pragma unroll
  for (int i = 0; i < 2; ++i)
#pragma unroll
    for (int j = 0; j < 2; ++j)
      store_c_frag(Cs + (mBase + 16 * i) * 66 + nBase + 16 * j, 66, c[i][j]);
  __syncthreads();

  const int bI = m0 >> 11;           // batch (tile never crosses batch: 2048%128==0)
  const int s0 = m0 & (Sc - 1);
  if (z < 2) {
    // bias + RoPE; write [B,H,S,DH]
    for (int e = 0; e < 32; ++e) {
      int lin = e * 256 + tid;
      int row = lin >> 6, col = lin & 63;
      if (col < 32) {
        int s = s0 + row;
        float x1 = Cs[row * 66 + col] + bias[n0h + col];
        float x2 = Cs[row * 66 + col + 32] + bias[n0h + col + 32];
        float t  = ts[bI * Sc + s];
        float invf = __powf(10000.f, -(float)col / 32.f);
        float sn, cs;
        __sincosf(t * invf, &sn, &cs);
        float o1 = x1 * cs - x2 * sn;
        float o2 = x1 * sn + x2 * cs;
        if (z == 0) { o1 *= 0.125f; o2 *= 0.125f; }  // fold 1/sqrt(64)
        _Float16* out = (z == 0) ? Qh : Kh;
        size_t baseo = ((size_t)(bI * Hh + h) * Sc + s) * DHc;
        out[baseo + col]      = (_Float16)o1;
        out[baseo + col + 32] = (_Float16)o2;
      }
    }
  } else {
    // V: bias, write transposed [B,H,DH,S] (coalesced: s contiguous per d)
    for (int e = 0; e < 32; ++e) {
      int lin = e * 256 + tid;
      int d = lin >> 7, srow = lin & 127;           // 64 d x 128 s
      float v = Cs[srow * 66 + d] + bias[n0h + d];
      Vt[((size_t)(bI * Hh + h) * DHc + d) * Sc + s0 + srow] = (_Float16)v;
    }
  }
}

// ---------------------------------------------------------------------------
// Flash attention: grid = (S/64, B*H). 64 query rows per block, stream keys.
// Q,K: [B,H,S,DH]; V: [B,H,DH,S] (dh-major for N-major B fragments).
// ---------------------------------------------------------------------------
__global__ __launch_bounds__(256) void attn_kernel(
    const _Float16* __restrict__ Qh, const _Float16* __restrict__ Kh,
    const _Float16* __restrict__ Vt, const int* __restrict__ mask,
    _Float16* __restrict__ Oh) {
  __shared__ _Float16 Qs[64 * 72];   // [q][dh]   row-major A
  __shared__ _Float16 Ks[64 * 72];   // [key][dh] == N-major B for Q*K^T
  __shared__ _Float16 Vs[64 * 72];   // [dh][key] == N-major B for P~*V
  __shared__ _Float16 Ph[64 * 72];   // P~ row-major A
  __shared__ float    Pf[64 * 66];
  __shared__ float    mrow[64], lrow[64], srow[64], maskb[64];

  const int bh = blockIdx.y;  // b*H + h
  const int b  = bh >> 3, h = bh & 7;
  const int q0 = blockIdx.x * 64;
  const size_t base  = (size_t)bh * Sc * DHc;  // Q/K base
  const size_t baseV = (size_t)bh * DHc * Sc;  // V^T base
  const int tid = threadIdx.x, wid = tid >> 5;
  const int mTile = (wid & 3) * 16, nHalf = (wid >> 2) * 32;

  {  // stage Q block (async)
    int row = tid >> 2, seg = (tid & 3) * 16;
    async_copy16h(Qs + row * 72 + seg,
                  Qh + base + (size_t)(q0 + row) * DHc + seg);
  }
  if (tid < 64) { mrow[tid] = -1e30f; lrow[tid] = 0.f; }

  v8f o[2];
  o[0] = (v8f){}; o[1] = (v8f){};
  async_wait();
  __syncthreads();

  for (int kb = 0; kb < Sc / 64; ++kb) {
    const int k0 = kb * 64;
    {  // stage K rows and V^T rows async (both N-major targets)
      int row = tid >> 2, seg = (tid & 3) * 16;
      async_copy16h(Ks + row * 72 + seg,
                    Kh + base + (size_t)(k0 + row) * DHc + seg);
      async_copy16h(Vs + row * 72 + seg,
                    Vt + baseV + (size_t)row * Sc + k0 + seg);
    }
    if (tid < 64) maskb[tid] = mask[b * Sc + k0 + tid] ? 0.f : -1e9f;
    async_wait();
    __syncthreads();

    // P = Q * K^T  (64x64, K-dim = DH = 64 -> two wmma k-steps)
    v8f p[2];
    p[0] = (v8f){}; p[1] = (v8f){};
#pragma unroll
    for (int ks = 0; ks < DHc; ks += 32) {
      v16h a = load_a_frag(Qs + mTile * 72 + ks, 72);
#pragma unroll
      for (int j = 0; j < 2; ++j) {
        v16h bt = load_b_frag_nmajor(Ks + (nHalf + 16 * j) * 72 + ks, 72);
        p[j] = wmma_f16(a, bt, p[j]);
      }
    }
    {  // scatter logits + mask bias into LDS
      int lane = tid & 31, col = lane & 15, ro = (lane >> 4) << 3;
#pragma unroll
      for (int j = 0; j < 2; ++j)
#pragma unroll
        for (int r = 0; r < 8; ++r) {
          int cc = nHalf + 16 * j + col;
          Pf[(mTile + ro + r) * 66 + cc] = p[j][r] + maskb[cc];
        }
    }
    __syncthreads();

    // online softmax per row (one thread per query row)
    if (tid < 64) {
      float mo = mrow[tid], mx = mo;
      for (int cc = 0; cc < 64; ++cc) mx = fmaxf(mx, Pf[tid * 66 + cc]);
      float sc = __expf(mo - mx);
      float l  = lrow[tid] * sc;
      for (int cc = 0; cc < 64; ++cc) {
        float ev = __expf(Pf[tid * 66 + cc] - mx);
        Ph[tid * 72 + cc] = (_Float16)ev;
        l += ev;
      }
      mrow[tid] = mx; lrow[tid] = l; srow[tid] = sc;
    }
    __syncthreads();

    // rescale accumulator by exp(m_old - m_new)
    {
      int lane = tid & 31, ro = (lane >> 4) << 3;
#pragma unroll
      for (int j = 0; j < 2; ++j)
#pragma unroll
        for (int r = 0; r < 8; ++r) o[j][r] *= srow[mTile + ro + r];
    }

    // O += P~ * V  (K-dim = 64 keys -> two wmma k-steps)
#pragma unroll
    for (int ks = 0; ks < 64; ks += 32) {
      v16h a = load_a_frag(Ph + mTile * 72 + ks, 72);
#pragma unroll
      for (int j = 0; j < 2; ++j) {
        v16h bv = load_b_frag_nmajor(Vs + (nHalf + 16 * j) * 72 + ks, 72);
        o[j] = wmma_f16(a, bv, o[j]);
      }
    }
    __syncthreads();  // protect Ks/Vs/Ph/Pf before next block
  }

  // finalize: O /= l, write [B, S, H*DH] f16
#pragma unroll
  for (int j = 0; j < 2; ++j)
    store_c_frag(Pf + mTile * 66 + nHalf + 16 * j, 66, o[j]);
  __syncthreads();
  for (int e = 0; e < 16; ++e) {
    int lin = e * 256 + tid;
    int row = lin >> 6, col = lin & 63;
    float v = Pf[row * 66 + col] / lrow[row];
    Oh[((size_t)b * Sc + q0 + row) * HD + h * DHc + col] = (_Float16)v;
  }
}

// ---------------------------------------------------------------------------
// Output projection: out = attn @ Wo + bo (f32 out). Wo pre-transposed [N][K].
// grid = (64, 8).
// ---------------------------------------------------------------------------
__global__ __launch_bounds__(256) void oproj_kernel(
    const _Float16* __restrict__ Ah, const _Float16* __restrict__ Wot,
    const float* __restrict__ bo, float* __restrict__ out) {
  __shared__ _Float16 As[128 * 40];
  __shared__ _Float16 Bs[64 * 40];   // N-major
  __shared__ float    Cs[128 * 66];

  const int m0 = blockIdx.x * 128, n0 = blockIdx.y * 64;
  const int tid = threadIdx.x, wid = tid >> 5;
  const int mBase = (wid & 3) * 32, nBase = (wid >> 2) * 32;

  v8f c[2][2];
#pragma unroll
  for (int i = 0; i < 2; ++i)
#pragma unroll
    for (int j = 0; j < 2; ++j) c[i][j] = (v8f){};

  for (int kk = 0; kk < HD; kk += 32) {
    {
      int row = tid >> 1, seg = (tid & 1) * 16;
      async_copy16h(As + row * 40 + seg,
                    Ah + (size_t)(m0 + row) * HD + kk + seg);
      if (kk + 32 < HD)
        __builtin_prefetch(Ah + (size_t)(m0 + row) * HD + kk + 32 + seg, 0, 0);
    }
    {
      int n = tid >> 2, seg = (tid & 3) * 8;
      async_copy8h(Bs + n * 40 + seg, Wot + (size_t)(n0 + n) * HD + kk + seg);
    }
    async_wait();
    __syncthreads();
    v16h a[2], b[2];
#pragma unroll
    for (int i = 0; i < 2; ++i) a[i] = load_a_frag(As + (mBase + 16 * i) * 40, 40);
#pragma unroll
    for (int j = 0; j < 2; ++j) b[j] = load_b_frag_nmajor(Bs + (nBase + 16 * j) * 40, 40);
#pragma unroll
    for (int i = 0; i < 2; ++i)
#pragma unroll
      for (int j = 0; j < 2; ++j) c[i][j] = wmma_f16(a[i], b[j], c[i][j]);
    __syncthreads();
  }
#pragma unroll
  for (int i = 0; i < 2; ++i)
#pragma unroll
    for (int j = 0; j < 2; ++j)
      store_c_frag(Cs + (mBase + 16 * i) * 66 + nBase + 16 * j, 66, c[i][j]);
  __syncthreads();
  for (int e = 0; e < 32; ++e) {
    int lin = e * 256 + tid;
    int row = lin >> 6, col = lin & 63;
    out[(size_t)(m0 + row) * Dm + n0 + col] = Cs[row * 66 + col] + bo[n0 + col];
  }
}

// ---------------------------------------------------------------------------
// Launch
// ---------------------------------------------------------------------------
extern "C" void kernel_launch(void* const* d_in, const int* in_sizes, int n_in,
                              void* d_out, int out_size, void* d_ws,
                              size_t ws_size, hipStream_t stream) {
  const float* x  = (const float*)d_in[0];
  const float* ts = (const float*)d_in[1];
  const int*   mk = (const int*)d_in[2];
  const float* Wq = (const float*)d_in[3];
  const float* bq = (const float*)d_in[4];
  const float* Wk = (const float*)d_in[5];
  const float* bk = (const float*)d_in[6];
  const float* Wv = (const float*)d_in[7];
  const float* bv = (const float*)d_in[8];
  const float* Wo = (const float*)d_in[9];
  const float* bo = (const float*)d_in[10];
  float* out = (float*)d_out;

  // Workspace carve-up (f16 buffers)
  char* w = (char*)d_ws;
  _Float16* xh  = (_Float16*)w;  w += (size_t)Mrows * Dm * 2;
  _Float16* Wqt = (_Float16*)w;  w += (size_t)Dm * HD * 2;
  _Float16* Wkt = (_Float16*)w;  w += (size_t)Dm * HD * 2;
  _Float16* Wvt = (_Float16*)w;  w += (size_t)Dm * HD * 2;
  _Float16* Wot = (_Float16*)w;  w += (size_t)HD * Dm * 2;
  _Float16* Qh  = (_Float16*)w;  w += (size_t)Bc * Hh * Sc * DHc * 2;
  _Float16* Kh  = (_Float16*)w;  w += (size_t)Bc * Hh * Sc * DHc * 2;
  _Float16* Vt  = (_Float16*)w;  w += (size_t)Bc * Hh * DHc * Sc * 2;
  _Float16* Ah  = (_Float16*)w;  w += (size_t)Mrows * HD * 2;

  const int nx = Mrows * Dm;  // 4,194,304
  cvt_f32_f16_kernel<<<(nx + 255) / 256, 256, 0, stream>>>(x, xh, nx);
  dim3 tg(Dm / 32, Dm / 32);
  cvt_transpose_kernel<<<tg, 256, 0, stream>>>(Wq, Wqt);
  cvt_transpose_kernel<<<tg, 256, 0, stream>>>(Wk, Wkt);
  cvt_transpose_kernel<<<tg, 256, 0, stream>>>(Wv, Wvt);
  cvt_transpose_kernel<<<tg, 256, 0, stream>>>(Wo, Wot);

  qkv_rope_kernel<<<dim3(Mrows / 128, Hh, 3), 256, 0, stream>>>(
      xh, Wqt, Wkt, Wvt, bq, bk, bv, ts, Qh, Kh, Vt);

  attn_kernel<<<dim3(Sc / 64, Bc * Hh), 256, 0, stream>>>(Qh, Kh, Vt, mk, Ah);

  oproj_kernel<<<dim3(Mrows / 128, Dm / 64), 256, 0, stream>>>(Ah, Wot, bo, out);
}
